// Model_55920474194562
// MI455X (gfx1250) — matrix operand order repaired and checked by reading, not compiled, not run on hardware
//
#include <hip/hip_runtime.h>

#define B_ 64
#define N_ 100
#define H_ 128
#define E_ 128

typedef __attribute__((ext_vector_type(16))) __bf16 v16bf;
typedef __attribute__((ext_vector_type(8)))  __bf16 v8bf;
typedef __attribute__((ext_vector_type(8)))  float  v8f;

typedef __attribute__((ext_vector_type(4))) unsigned tdm_v4u;
typedef __attribute__((ext_vector_type(8))) int      tdm_v8i;
typedef __attribute__((ext_vector_type(4))) int      tdm_v4i;

#define FSYNC() do { __threadfence(); __syncthreads(); __threadfence(); } while (0)

__device__ __forceinline__ float sigmoidf_(float x) { return 1.0f / (1.0f + __expf(-x)); }

__device__ __forceinline__ unsigned pcg_hash(unsigned x) {
  x = x * 747796405u + 2891336453u;
  x = ((x >> ((x >> 28) + 4u)) ^ x) * 277803737u;
  return (x >> 22) ^ x;
}
__device__ __forceinline__ float rng01(unsigned seed, unsigned a, unsigned b, unsigned c) {
  unsigned h = pcg_hash(seed ^ (a * 2654435761u) ^ (b * 97062133u) ^ (c * 40503u + 0x9E3779B9u));
  return ((float)(h >> 8) + 0.5f) * (1.0f / 16777216.0f);
}

// ---- CDNA5 async global->LDS (ASYNCcnt) helpers --------------------------
__device__ __forceinline__ void async_load_b128(unsigned lds_off, const void* gptr) {
  asm volatile("global_load_async_to_lds_b128 %0, %1, off"
               :: "v"(lds_off), "v"((unsigned long long)gptr)
               : "memory");
}
__device__ __forceinline__ void wait_async0() {
#if __has_builtin(__builtin_amdgcn_s_wait_asynccnt)
  __builtin_amdgcn_s_wait_asynccnt(0);
#else
  asm volatile("s_wait_asynccnt 0" ::: "memory");
#endif
}
__device__ __forceinline__ void wait_tensor0() {
#if __has_builtin(__builtin_amdgcn_s_wait_tensorcnt)
  __builtin_amdgcn_s_wait_tensorcnt(0);
#else
  asm volatile("s_wait_tensorcnt 0" ::: "memory");
#endif
}

// ---- shared WMMA tile helper: C16x16 tile of (A[64x128] f32) @ (Bt[N][128] bf16) ----
__device__ __forceinline__ void wmma_tile(const float* __restrict__ Asrc,
                                          const __bf16* __restrict__ Bt,
                                          float* __restrict__ Cdst, int Ncols,
                                          int mt, int nt, int g, int c) {
  v8f acc = {};
  for (int k0 = 0; k0 < 128; k0 += 32) {
    const float4* p0 = (const float4*)&Asrc[(mt * 16 + c) * 128 + k0 + (g << 3)];
    const float4* p1 = (const float4*)&Asrc[(mt * 16 + c) * 128 + k0 + 16 + (g << 3)];
    float4 a0 = p0[0], a1 = p0[1], a2 = p1[0], a3 = p1[1];
    float af[16] = {a0.x, a0.y, a0.z, a0.w, a1.x, a1.y, a1.z, a1.w,
                    a2.x, a2.y, a2.z, a2.w, a3.x, a3.y, a3.z, a3.w};
    v16bf a, bm;
    #pragma unroll
    for (int i = 0; i < 16; ++i) a[i] = (__bf16)af[i];
    const __bf16* w = Bt + (nt * 16 + c) * 128 + k0 + (g << 3);
    v8bf lo = *(const v8bf*)w, hi = *(const v8bf*)(w + 16);
    #pragma unroll
    for (int i = 0; i < 8; ++i) { bm[i] = lo[i]; bm[i + 8] = hi[i]; }
    acc = __builtin_amdgcn_wmma_f32_16x16x32_bf16(false, a, false, bm, (short)0, acc, false, false);
  }
  #pragma unroll
  for (int r = 0; r < 8; ++r)
    Cdst[(mt * 16 + r + g * 8) * Ncols + nt * 16 + c] = acc[r];
}

// ---------------- node embedding -------------------------------------------
__global__ __launch_bounds__(256) void k_embed(const float* __restrict__ node,
                                               const float* __restrict__ demand,
                                               const float* __restrict__ Wn0,
                                               const float* __restrict__ bn0,
                                               float* __restrict__ h) {
  int idx = blockIdx.x * blockDim.x + threadIdx.x;
  if (idx >= B_ * N_ * H_) return;
  int j = idx & (H_ - 1);
  int bn = idx >> 7;
  float x = node[bn * 2 + 0], y = node[bn * 2 + 1], d = demand[bn];
  float v = x * Wn0[j] + y * Wn0[H_ + j] + d * Wn0[2 * H_ + j] + bn0[j];
  h[idx] = fmaxf(v, 0.0f);
}

// ---- pack GCN weights -> bf16 transposed [n][k] ---------------------------
__global__ __launch_bounds__(256) void k_pack_weights(const float* __restrict__ Wn,
                                                      const float* __restrict__ Wsrc,
                                                      __bf16* __restrict__ WtN,
                                                      __bf16* __restrict__ WtS) {
  int idx = blockIdx.x * blockDim.x + threadIdx.x;
  if (idx >= 3 * H_ * H_) return;
  int l = idx >> 14, r = idx & 16383;
  int n = r >> 7, k = r & 127;
  WtN[l * H_ * H_ + n * H_ + k] = (__bf16)Wn[l * H_ * H_ + k * H_ + n];
  WtS[l * H_ * H_ + n * H_ + k] = (__bf16)Wsrc[l * H_ * H_ + k * H_ + n];
}

// ---- pack decoder weights -> bf16 transposed [out][k] ---------------------
__global__ __launch_bounds__(256) void k_pack_decoder(const float* __restrict__ Wih,
                                                      const float* __restrict__ Whh,
                                                      const float* __restrict__ Wq,
                                                      __bf16* __restrict__ WihT,
                                                      __bf16* __restrict__ WhhT,
                                                      __bf16* __restrict__ WqT) {
  int idx = blockIdx.x * blockDim.x + threadIdx.x;
  if (idx < 2 * 384 * 128) {
    int l = idx / 49152, r = idx % 49152;
    int j = r >> 7, k = r & 127;                       // j: gate col 0..383
    WihT[l * 49152 + j * 128 + k] = (__bf16)Wih[l * 49152 + k * 384 + j];
    WhhT[l * 49152 + j * 128 + k] = (__bf16)Whh[l * 49152 + k * 384 + j];
  }
  if (idx < H_ * H_) {
    int n = idx >> 7, k = idx & 127;
    WqT[n * 128 + k] = (__bf16)Wq[k * 128 + n];
  }
}

// ---------------- A = softmax(-dis) row-wise, one wave per row -------------
__global__ __launch_bounds__(256) void k_softmax_rows(const float* __restrict__ dis,
                                                      float* __restrict__ A) {
  int wave = threadIdx.x >> 5, lane = threadIdx.x & 31;
  int row = blockIdx.x * 8 + wave;
  const float* src = dis + (long)row * N_;
  float vals[4];
  float vmax = -3.4e38f;
  #pragma unroll
  for (int t = 0; t < 4; ++t) {
    int j = lane + 32 * t;
    vals[t] = (j < N_) ? -src[j] : -3.4e38f;
    vmax = fmaxf(vmax, vals[t]);
  }
  #pragma unroll
  for (int off = 16; off > 0; off >>= 1) vmax = fmaxf(vmax, __shfl_xor(vmax, off, 32));
  float sum = 0.0f;
  #pragma unroll
  for (int t = 0; t < 4; ++t) {
    int j = lane + 32 * t;
    vals[t] = (j < N_) ? __expf(vals[t] - vmax) : 0.0f;
    sum += vals[t];
  }
  #pragma unroll
  for (int off = 16; off > 0; off >>= 1) sum += __shfl_xor(sum, off, 32);
  float inv = 1.0f / sum;
  #pragma unroll
  for (int t = 0; t < 4; ++t) {
    int j = lane + 32 * t;
    if (j < N_) A[(long)row * N_ + j] = vals[t] * inv;
  }
}

// ------ dual GEMM: T1t(bf16,transposed) = h @ Wn, T2 = h @ Ws --------------
// h tile staged via async global->LDS; weights bf16 transposed (2x b128/frag).
// T1t written as [b][n][kpad=128] bf16 so the neighbor GEMM's B fragments are
// contiguous 32B loads with zero-padded K (pads pre-zeroed once per launch).
__global__ __launch_bounds__(256) void k_gcn_gemm(const float* __restrict__ h,
                                                  const __bf16* __restrict__ Wtn,
                                                  const __bf16* __restrict__ Wts,
                                                  __bf16* __restrict__ T1t,
                                                  float* __restrict__ T2) {
  __shared__ float ldsH[16 * H_];
  int wave = threadIdx.x >> 5, lane = threadIdx.x & 31;
  int m0 = blockIdx.x * 16;
  int n0 = wave * 16;
  int g = lane >> 4;
  int c = lane & 15;

  for (int ch = threadIdx.x; ch < 16 * 32; ch += 256) {
    int r = ch >> 5, c4 = (ch & 31) << 2;
    unsigned loff = (unsigned)(unsigned long long)&ldsH[r * H_ + c4];
    async_load_b128(loff, h + ((long)(m0 + r) * H_ + c4));
  }
  wait_async0();
  __syncthreads();

  v8f acc1 = {};
  v8f acc2 = {};
  for (int k0 = 0; k0 < H_; k0 += 32) {
    const float4* hp0 = (const float4*)&ldsH[c * H_ + k0 + (g << 3)];
    const float4* hp1 = (const float4*)&ldsH[c * H_ + k0 + 16 + (g << 3)];
    float4 a0 = hp0[0], a1 = hp0[1], a2 = hp1[0], a3 = hp1[1];
    float af[16] = {a0.x, a0.y, a0.z, a0.w, a1.x, a1.y, a1.z, a1.w,
                    a2.x, a2.y, a2.z, a2.w, a3.x, a3.y, a3.z, a3.w};
    v16bf a, b1, b2;
    #pragma unroll
    for (int i = 0; i < 16; ++i) a[i] = (__bf16)af[i];

    const __bf16* w1 = Wtn + (n0 + c) * H_ + k0 + (g << 3);
    const __bf16* w2 = Wts + (n0 + c) * H_ + k0 + (g << 3);
    if (k0 + 32 < H_) {
      __builtin_prefetch(w1 + 32, 0, 1);
      __builtin_prefetch(w2 + 32, 0, 1);
    }
    v8bf l1 = *(const v8bf*)w1, h1 = *(const v8bf*)(w1 + 16);
    v8bf l2 = *(const v8bf*)w2, h2 = *(const v8bf*)(w2 + 16);
    #pragma unroll
    for (int i = 0; i < 8; ++i) { b1[i] = l1[i]; b1[i + 8] = h1[i]; b2[i] = l2[i]; b2[i + 8] = h2[i]; }

    acc1 = __builtin_amdgcn_wmma_f32_16x16x32_bf16(false, a, false, b1, (short)0, acc1, false, false);
    acc2 = __builtin_amdgcn_wmma_f32_16x16x32_bf16(false, a, false, b2, (short)0, acc2, false, false);
  }
  #pragma unroll
  for (int r = 0; r < 8; ++r) {
    int mm = m0 + r + g * 8;
    int nn = n0 + c;
    int bb = mm / 100, rr = mm % 100;
    T1t[((long)((bb << 7) + nn) << 7) + rr] = (__bf16)acc1[r];
    T2[(long)mm * H_ + nn] = acc2[r];
  }
}

// ------ neighbor aggregate: h = relu(T2 + A_b @ T1_b) ----------------------
// A tile staged via TDM tensor_load_to_lds (zero-fill OOB rows); B fragments
// from bf16-transposed T1t: 2x global_load_b128, no converts, no guards.
__global__ __launch_bounds__(256) void k_gcn_neigh(const float* __restrict__ A,
                                                   const __bf16* __restrict__ T1t,
                                                   const float* __restrict__ T2,
                                                   float* __restrict__ hout) {
  __shared__ float ldsA[16 * N_];
  int b = blockIdx.x / 7, mt = blockIdx.x % 7;
  int wave = threadIdx.x >> 5, lane = threadIdx.x & 31;
  int n0 = wave * 16, m0 = mt * 16;
  int g = lane >> 4, c = lane & 15;

  for (int i = threadIdx.x; i < 16 * N_; i += 256) ldsA[i] = 0.0f;
  __syncthreads();

#if __has_builtin(__builtin_amdgcn_tensor_load_to_lds)
  if (threadIdx.x < 32) {
    unsigned long long ga = (unsigned long long)(A + ((long)b * N_ + m0) * N_);
    int rows = N_ - m0; if (rows > 16) rows = 16;
    tdm_v4u g0 = {};
    g0.x = 1u;
    g0.y = (unsigned)(unsigned long long)&ldsA[0];
    g0.z = (unsigned)(ga & 0xffffffffu);
    g0.w = (unsigned)((ga >> 32) & 0x01ffffffu) | (2u << 30);
    tdm_v8i g1 = {};
    g1[0] = (int)(2u << 16);                 // data_size = 4 bytes
    g1[1] = (int)(100u << 16);               // tensor_dim0 = 100
    g1[2] = (int)((unsigned)rows << 16);     // tensor_dim1 (OOB rows -> 0)
    g1[3] = (int)(100u << 16);               // tile_dim0 = 100
    g1[4] = 16;                              // tile_dim1 = 16
    g1[5] = 100;                             // tensor_dim0_stride = 100
    tdm_v4i g2 = {}; tdm_v4i g3 = {};
#if __clang_major__ >= 23
    tdm_v8i g4 = {};
    __builtin_amdgcn_tensor_load_to_lds(g0, g1, g2, g3, g4, 0);
#else
    __builtin_amdgcn_tensor_load_to_lds(g0, g1, g2, g3, 0);
#endif
    wait_tensor0();
  }
#else
  for (int ch = threadIdx.x; ch < 16 * 25; ch += 256) {
    int r = ch / 25, c4 = (ch % 25) * 4;
    if (m0 + r < N_) {
      unsigned loff = (unsigned)(unsigned long long)&ldsA[r * N_ + c4];
      async_load_b128(loff, A + ((long)b * N_ + m0 + r) * N_ + c4);
    }
  }
  wait_async0();
#endif
  __syncthreads();

  v8f acc = {};
  for (int k0 = 0; k0 < 128; k0 += 32) {
    v16bf a, bm;
    #pragma unroll
    for (int i = 0; i < 16; ++i) {
      int ka = k0 + (g << 4) + i;
      a[i] = (__bf16)((ka < N_) ? ldsA[c * N_ + ka] : 0.0f);
    }
    const __bf16* tb = T1t + ((long)((b << 7) + n0 + c) << 7) + k0 + (g << 4);
    v8bf lo = *(const v8bf*)tb, hi = *(const v8bf*)(tb + 8);
    #pragma unroll
    for (int i = 0; i < 8; ++i) { bm[i] = lo[i]; bm[i + 8] = hi[i]; }
    acc = __builtin_amdgcn_wmma_f32_16x16x32_bf16(false, a, false, bm, (short)0, acc, false, false);
  }
  #pragma unroll
  for (int r = 0; r < 8; ++r) {
    int mm = m0 + r + g * 8;
    if (mm < N_) {
      long idx = ((long)b * N_ + mm) * H_ + n0 + c;
      hout[idx] = fmaxf(T2[idx] + acc[r], 0.0f);
    }
  }
}

// ---------------- persistent decoder with WMMA GRUs ------------------------
// 16 waves; per step: gi/gh = X@WihT, H@WhhT (96 tile jobs x2), gates,
// second GRU, q = h1@WqT (32 tiles), logits (VALU batched dot), select.
__global__ __launch_bounds__(512) void k_decode(const float* __restrict__ x,
                                                const __bf16* __restrict__ WihT,
                                                const __bf16* __restrict__ WhhT,
                                                const __bf16* __restrict__ WqT,
                                                const float* __restrict__ bih,
                                                const float* __restrict__ bhh,
                                                int sample, unsigned seed,
                                                float* __restrict__ ws,
                                                int* __restrict__ last,
                                                int* __restrict__ sol,
                                                float* __restrict__ lp_out) {
  const int T = blockDim.x;
  const int t = threadIdx.x;
  const int wid = t >> 5, lane = t & 31;
  const int g = lane >> 4, c = lane & 15;
  float* emb    = ws;
  float* hid0   = ws + 8192;
  float* hid1   = ws + 16384;
  float* nh0    = ws + 24576;
  float* nh1    = ws + 32768;
  float* q      = ws + 40960;
  float* logits = ws + 49152;           // 6400
  float* gi     = ws + 55552;           // 64x384
  float* gh     = ws + 80128;           // 64x384

  for (int i = t; i < B_ * H_; i += T) { hid0[i] = 0.0f; hid1[i] = 0.0f; }
  for (int i = t; i < B_; i += T) { last[i] = 0; sol[i * N_] = 0; }
  FSYNC();

  const float scale = 0.08838834764831845f;   // 1/sqrt(128)
  for (int step = 0; step < N_ - 1; ++step) {
    for (int i = t; i < B_ * H_; i += T) {
      int b = i >> 7, j = i & (H_ - 1);
      emb[i] = x[((long)b * N_ + last[b]) * H_ + j];
    }
    FSYNC();
    // ---- GRU layer 0: gi = emb@Wih0, gh = hid0@Whh0 (WMMA tiles) ----
    for (int job = wid; job < 96; job += 16) {
      int mt = job & 3, nt = job >> 2;
      wmma_tile(emb,  WihT, gi, 384, mt, nt, g, c);
      wmma_tile(hid0, WhhT, gh, 384, mt, nt, g, c);
    }
    FSYNC();
    for (int i = t; i < B_ * H_; i += T) {
      int b = i >> 7, j = i & (H_ - 1);
      const float* gib = gi + b * 384;
      const float* ghb = gh + b * 384;
      float r = sigmoidf_(gib[j] + bih[j] + ghb[j] + bhh[j]);
      float z = sigmoidf_(gib[128 + j] + bih[128 + j] + ghb[128 + j] + bhh[128 + j]);
      float n = tanhf(gib[256 + j] + bih[256 + j] + r * (ghb[256 + j] + bhh[256 + j]));
      nh0[i] = (1.0f - z) * n + z * hid0[i];
    }
    FSYNC();
    // ---- GRU layer 1: gi = nh0@Wih1, gh = hid1@Whh1 ----
    for (int job = wid; job < 96; job += 16) {
      int mt = job & 3, nt = job >> 2;
      wmma_tile(nh0,  WihT + 49152, gi, 384, mt, nt, g, c);
      wmma_tile(hid1, WhhT + 49152, gh, 384, mt, nt, g, c);
    }
    FSYNC();
    for (int i = t; i < B_ * H_; i += T) {
      int b = i >> 7, j = i & (H_ - 1);
      const float* gib = gi + b * 384;
      const float* ghb = gh + b * 384;
      float r = sigmoidf_(gib[j] + bih[384 + j] + ghb[j] + bhh[384 + j]);
      float z = sigmoidf_(gib[128 + j] + bih[512 + j] + ghb[128 + j] + bhh[512 + j]);
      float n = tanhf(gib[256 + j] + bih[640 + j] + r * (ghb[256 + j] + bhh[640 + j]));
      nh1[i] = (1.0f - z) * n + z * hid1[i];
    }
    FSYNC();
    // ---- q = nh1 @ Wq (WMMA, 32 tiles) ----
    for (int job = wid; job < 32; job += 16) {
      int mt = job & 3, nt = job >> 2;
      wmma_tile(nh1, WqT, q, 128, mt, nt, g, c);
    }
    FSYNC();
    for (int i = t; i < B_ * N_; i += T) {
      int b = i / N_, n = i % N_;
      const float* qb = q + b * H_;
      const float* xb = x + ((long)b * N_ + n) * H_;
      float s = 0.0f;
      for (int k = 0; k < H_; ++k) s += qb[k] * xb[k];
      logits[i] = s * scale;
    }
    FSYNC();
    if (t < B_) {
      const float* lg = logits + t * N_;
      float m = -3.4e38f;
      for (int j = 0; j < N_; ++j) m = fmaxf(m, lg[j]);
      float sum = 0.0f;
      for (int j = 0; j < N_; ++j) sum += __expf(lg[j] - m);
      float lse = m + __logf(sum);
      int ind = 0;
      if (sample) {
        float best = -3.4e38f;
        for (int j = 0; j < N_; ++j) {
          float u = rng01(seed, (unsigned)step, (unsigned)t, (unsigned)j);
          float gmb = lg[j] - __logf(-__logf(u));
          if (gmb > best) { best = gmb; ind = j; }
        }
      } else {
        float best = -3.4e38f;
        for (int j = 0; j < N_; ++j) if (lg[j] > best) { best = lg[j]; ind = j; }
      }
      if (sample && lp_out) lp_out[t * (N_ - 1) + step] = lg[ind] - lse;
      last[t] = ind;
      sol[t * N_ + step + 1] = ind;
    }
    FSYNC();
    for (int i = t; i < B_ * H_; i += T) { hid0[i] = nh0[i]; hid1[i] = nh1[i]; }
    FSYNC();
  }
}

// ---------------- tour length from solution --------------------------------
__global__ void k_distance(const float* __restrict__ dis, const int* __restrict__ sol,
                           float* __restrict__ out) {
  int b = blockIdx.x * blockDim.x + threadIdx.x;
  if (b >= B_) return;
  float s = 0.0f;
  for (int tt = 0; tt < N_ - 1; ++tt)
    s += dis[(long)b * N_ * N_ + sol[b * N_ + tt] * N_ + sol[b * N_ + tt + 1]];
  out[b] = s;
}

// ------ fused edge head: softmax(relu(dis*We+be) @ Wc + bc) — no 328MB e ---
__global__ __launch_bounds__(256) void k_predict(const float* __restrict__ dis,
                                                 const float* __restrict__ We,
                                                 const float* __restrict__ be,
                                                 const float* __restrict__ Wc,
                                                 const float* __restrict__ bc,
                                                 float* __restrict__ out) {
  int idx = blockIdx.x * blockDim.x + threadIdx.x;
  if (idx >= B_ * N_ * N_) return;
  float d = dis[idx];
  float s0 = bc[0], s1 = bc[1];
  for (int k = 0; k < E_; ++k) {
    float e = fmaxf(d * We[k] + be[k], 0.0f);
    s0 += e * Wc[2 * k + 0];
    s1 += e * Wc[2 * k + 1];
  }
  float m = fmaxf(s0, s1);
  float e0 = __expf(s0 - m), e1 = __expf(s1 - m);
  float inv = 1.0f / (e0 + e1);
  out[(long)idx * 2 + 0] = e0 * inv;
  out[(long)idx * 2 + 1] = e1 * inv;
}

__global__ void k_zero(float* __restrict__ p, int n) {
  int i = blockIdx.x * blockDim.x + threadIdx.x;
  if (i < n) p[i] = 0.0f;
}

__global__ void k_scatter(const int* __restrict__ sol, float* __restrict__ solmat) {
  int idx = blockIdx.x * blockDim.x + threadIdx.x;
  if (idx >= B_ * (N_ - 1)) return;
  int b = idx / (N_ - 1), tt = idx % (N_ - 1);
  int i = sol[b * N_ + tt], j = sol[b * N_ + tt + 1];
  solmat[(long)b * N_ * N_ + i * N_ + j] = 1.0f;
}

extern "C" void kernel_launch(void* const* d_in, const int* in_sizes, int n_in,
                              void* d_out, int out_size, void* d_ws, size_t ws_size,
                              hipStream_t stream) {
  (void)in_sizes; (void)n_in; (void)out_size; (void)ws_size;
  const float* node   = (const float*)d_in[0];
  const float* demand = (const float*)d_in[1];
  const float* dis    = (const float*)d_in[2];
  const float* Wn0    = (const float*)d_in[3];
  const float* bn0    = (const float*)d_in[4];
  const float* Ws     = (const float*)d_in[5];
  const float* Wngh   = (const float*)d_in[6];
  const float* We     = (const float*)d_in[7];
  const float* be     = (const float*)d_in[8];
  const float* Wih    = (const float*)d_in[9];
  const float* Whh    = (const float*)d_in[10];
  const float* bih    = (const float*)d_in[11];
  const float* bhh    = (const float*)d_in[12];
  const float* Wq     = (const float*)d_in[13];
  const float* Wc     = (const float*)d_in[14];
  const float* bc     = (const float*)d_in[15];

  float* out         = (float*)d_out;
  float* lp_sample   = out;                         // 64*99
  float* dist_sample = out + 6336;                  // 64
  float* dist_greedy = out + 6400;                  // 64
  float* predict     = out + 6464;                  // 64*100*100*2
  float* solmat      = out + 6464 + 1280000;        // 64*100*100

  float* ws  = (float*)d_ws;
  float* h   = ws;                                  // 819200
  float* Aad = ws + 819200;                         // 640000
  __bf16* T1t = (__bf16*)(ws + 1459200);            // 64*128*128 bf16 (524288 f)
  float* T2  = ws + 2278400;                        // 819200
  float* dec = ws + 3097600;                        // 104704 (emb..gh)
  int*   last  = (int*)(ws + 3202304);              // 64
  int*   sol_s = (int*)(ws + 3202368);              // 6400
  int*   sol_g = (int*)(ws + 3208768);              // 6400
  __bf16* WtN  = (__bf16*)(ws + 3215168);           // 3*128*128 bf16
  __bf16* WtS  = (__bf16*)(ws + 3239744);
  __bf16* WihT = (__bf16*)(ws + 3264320);           // 2*384*128 bf16
  __bf16* WhhT = (__bf16*)(ws + 3313472);
  __bf16* WqT  = (__bf16*)(ws + 3362624);           // 128*128 bf16

  k_embed<<<3200, 256, 0, stream>>>(node, demand, Wn0, bn0, h);
  k_pack_weights<<<192, 256, 0, stream>>>(Wngh, Ws, WtN, WtS);
  k_pack_decoder<<<384, 256, 0, stream>>>(Wih, Whh, Wq, WihT, WhhT, WqT);
  k_softmax_rows<<<800, 256, 0, stream>>>(dis, Aad);
  k_zero<<<2048, 256, 0, stream>>>((float*)T1t, 524288);   // zero K-pads of T1t
  for (int l = 0; l < 3; ++l) {
    k_gcn_gemm<<<400, 256, 0, stream>>>(h, WtN + l * H_ * H_, WtS + l * H_ * H_, T1t, T2);
    k_gcn_neigh<<<448, 256, 0, stream>>>(Aad, T1t, T2, h);
  }
  k_decode<<<1, 512, 0, stream>>>(h, WihT, WhhT, WqT, bih, bhh, 1, 0x5EEDu, dec, last, sol_s, lp_sample);
  k_distance<<<1, 64, 0, stream>>>(dis, sol_s, dist_sample);
  k_decode<<<1, 512, 0, stream>>>(h, WihT, WhhT, WqT, bih, bhh, 0, 0u, dec, last, sol_g, nullptr);
  k_distance<<<1, 64, 0, stream>>>(dis, sol_g, dist_greedy);
  k_predict<<<2500, 256, 0, stream>>>(dis, We, be, Wc, bc, predict);
  k_zero<<<2500, 256, 0, stream>>>(solmat, B_ * N_ * N_);
  k_scatter<<<25, 256, 0, stream>>>(sol_g, solmat);
}
